// sMLP_91336774517591
// MI455X (gfx1250) — matrix operand order repaired and checked by reading, not compiled, not run on hardware
//
#include <hip/hip_runtime.h>
#include <stdint.h>

// ---------------------------------------------------------------------------
// sMLP fused kernel for MI455X (gfx1250, wave32, WMMA)
//   y = relu(X @ W^T + b); out = m ? y : x;  m = (fg!=0) & (mask.T==0)
// GEMM: 131072x256 @ 256x256, bf16x3 split-precision via v_wmma_f32_16x16x32_bf16
// Roofline: 3x17.2 GFLOP of bf16 WMMA (hidden) under ~400 MB HBM traffic (~17us).
// ---------------------------------------------------------------------------

typedef __attribute__((ext_vector_type(16))) __bf16 v16bf;
typedef __attribute__((ext_vector_type(8)))  float  v8f;

#define N_TOK 16384
#define BATCH 8
#define CDIM  256
#define TTOT  (N_TOK * BATCH)     // 131072 tokens
#define WROW  264                 // 256 + 8 pad elems -> 528B row stride (bank-conflict free)

union BPack4 { __bf16 b[4]; uint2 u; };

__global__ __launch_bounds__(256, 1)
void smlp_wmma_kernel(const float* __restrict__ x,
                      const int*   __restrict__ mask,   // (B, N)
                      const int*   __restrict__ fg,     // (N, B) == flat token index
                      const float* __restrict__ W,      // (C, C) row-major [d][c]
                      const float* __restrict__ bias,   // (C,)
                      float*       __restrict__ out)    // x_new (T*C) then m (T)
{
    // W staged once per WG as bf16 hi + residual lo, padded rows. 132 KB LDS.
    __shared__ __align__(16) __bf16 sWhi[128 * WROW];
    __shared__ __align__(16) __bf16 sWlo[128 * WROW];

    const int tid  = threadIdx.x;
    const int lane = tid & 31;
    const int wave = tid >> 5;           // 0..7
    const int wgd  = blockIdx.x & 1;     // which 128-wide half of output channels
    const int wgt  = blockIdx.x >> 1;    // which 128-token block
    const int d0   = wgd * 128;
    const int tblk = wgt * 128;

    // ---- emit the m output (once, from the wgd==0 blocks) ----
    if (wgd == 0 && tid < 128) {
        const int t  = tblk + tid;
        const int nn = t >> 3, bb = t & 7;
        const float mv = ((fg[t] != 0) && (mask[bb * N_TOK + nn] == 0)) ? 1.0f : 0.0f;
        out[(size_t)TTOT * CDIM + t] = mv;
    }

    // ---- stage W[d0 .. d0+127][:] into LDS (native packed cvt, b64 stores) ----
    {
        const int r  = tid >> 1;            // 0..127
        const int cb = (tid & 1) * 128;     // 0 or 128
        const float4* wrow = (const float4*)(W + (size_t)(d0 + r) * CDIM + cb);
        __bf16* hi = sWhi + r * WROW + cb;
        __bf16* lo = sWlo + r * WROW + cb;
        #pragma unroll
        for (int j = 0; j < 32; ++j) {
            const float4 w4 = wrow[j];
            BPack4 uh, ul;
            uh.b[0] = (__bf16)w4.x;  uh.b[1] = (__bf16)w4.y;
            uh.b[2] = (__bf16)w4.z;  uh.b[3] = (__bf16)w4.w;
            ul.b[0] = (__bf16)(w4.x - (float)uh.b[0]);
            ul.b[1] = (__bf16)(w4.y - (float)uh.b[1]);
            ul.b[2] = (__bf16)(w4.z - (float)uh.b[2]);
            ul.b[3] = (__bf16)(w4.w - (float)uh.b[3]);
            *(uint2*)(hi + j * 4) = uh.u;
            *(uint2*)(lo + j * 4) = ul.u;
        }
    }
    __syncthreads();

    // ---- per-wave tile: 16 tokens x 128 output channels ----
    const int half = lane >> 4;          // 0/1
    const int nl   = lane & 15;
    const int t0   = tblk + wave * 16;
    const float* xrow = x + (size_t)(t0 + nl) * CDIM;   // A-fragment source row

    // token gate for the epilogue rows this lane owns: t = t0 + v + 8*half
    float mrow[8];
    #pragma unroll
    for (int v = 0; v < 8; ++v) {
        const int t  = t0 + v + 8 * half;
        const int nn = t >> 3, bb = t & 7;
        mrow[v] = ((fg[t] != 0) && (mask[bb * N_TOK + nn] == 0)) ? 1.0f : 0.0f;
    }

    v8f acc[8] = {};   // 8 tiles of 16x16 f32 (d-tiles dt=0..7)

    #pragma unroll
    for (int kt = 0; kt < 8; ++kt) {
        const int k0 = kt * 32;
        // A fragment (16x32 bf16): lane<16 holds K in {k0..k0+7} U {k0+16..k0+23}
        const float4 a0 = *(const float4*)(xrow + k0 + half * 8);
        const float4 a1 = *(const float4*)(xrow + k0 + half * 8 + 4);
        const float4 a2 = *(const float4*)(xrow + k0 + half * 8 + 16);
        const float4 a3 = *(const float4*)(xrow + k0 + half * 8 + 20);

        const float av[16] = {a0.x, a0.y, a0.z, a0.w,  a1.x, a1.y, a1.z, a1.w,
                              a2.x, a2.y, a2.z, a2.w,  a3.x, a3.y, a3.z, a3.w};
        v16bf ahi, alo;
        #pragma unroll
        for (int e = 0; e < 16; ++e) {
            const __bf16 h = (__bf16)av[e];         // v_cvt_pk_bf16_f32 pairs
            ahi[e] = h;
            alo[e] = (__bf16)(av[e] - (float)h);    // residual in bf16
        }

        #pragma unroll
        for (int dt = 0; dt < 8; ++dt) {
            // B fragment (32x16 bf16): lane holds N = nl, K = k0 + 8*half + {0..7, 16..23}
            const __bf16* bp_hi = sWhi + (dt * 16 + nl) * WROW + k0 + half * 8;
            const __bf16* bp_lo = sWlo + (dt * 16 + nl) * WROW + k0 + half * 8;
            union { uint4 q[2]; v16bf v; } bhi, blo;
            bhi.q[0] = *(const uint4*)(bp_hi);
            bhi.q[1] = *(const uint4*)(bp_hi + 16);
            blo.q[0] = *(const uint4*)(bp_lo);
            blo.q[1] = *(const uint4*)(bp_lo + 16);

            // bf16x3 split accumulation: Alo*Bhi + Ahi*Blo + Ahi*Bhi (f32 acc)
            acc[dt] = __builtin_amdgcn_wmma_f32_16x16x32_bf16(
                false, alo, false, bhi.v, (short)0, acc[dt], false, false);
            acc[dt] = __builtin_amdgcn_wmma_f32_16x16x32_bf16(
                false, ahi, false, blo.v, (short)0, acc[dt], false, false);
            acc[dt] = __builtin_amdgcn_wmma_f32_16x16x32_bf16(
                false, ahi, false, bhi.v, (short)0, acc[dt], false, false);
        }
    }

    // ---- epilogue: bias + relu + masked blend with original x ----
    // C/D layout: VGPR v, lanes 0-15 -> M=v, lanes 16-31 -> M=v+8; N = nl
    #pragma unroll
    for (int dt = 0; dt < 8; ++dt) {
        const int d  = d0 + dt * 16 + nl;
        const float bv = bias[d];
        #pragma unroll
        for (int v = 0; v < 8; ++v) {
            const int t = t0 + v + 8 * half;
            float y = acc[dt][v] + bv;
            y = fmaxf(y, 0.0f);
            const float xo = x[(size_t)t * CDIM + d];
            out[(size_t)t * CDIM + d] = (mrow[v] != 0.0f) ? y : xo;
        }
    }
}

extern "C" void kernel_launch(void* const* d_in, const int* in_sizes, int n_in,
                              void* d_out, int out_size, void* d_ws, size_t ws_size,
                              hipStream_t stream) {
    (void)in_sizes; (void)n_in; (void)out_size; (void)d_ws; (void)ws_size;
    const float* x    = (const float*)d_in[0];
    const int*   mask = (const int*)d_in[1];
    const int*   fg   = (const int*)d_in[2];
    const float* W    = (const float*)d_in[3];
    const float* b    = (const float*)d_in[4];
    float*       out  = (float*)d_out;

    dim3 grid(2048), block(256);   // 2048 WGs x (8 waves of 32): 128 tokens x 128 chans each
    hipLaunchKernelGGL(smlp_wmma_kernel, grid, block, 0, stream,
                       x, mask, fg, W, b, out);
}
